// FirstStageEncoderBlock_27754078667054
// MI455X (gfx1250) — matrix operand order repaired and checked by reading, not compiled, not run on hardware
//
#include <hip/hip_runtime.h>
#include <hip/hip_bf16.h>
#include <math.h>

typedef __attribute__((ext_vector_type(16))) _Float16 v16h;
typedef __attribute__((ext_vector_type(8)))  _Float16 v8h;
typedef __attribute__((ext_vector_type(8)))  float    v8f;

namespace {
constexpr int kB  = 2;
constexpr int kT  = 6;
constexpr int kC  = 64;     // COUT
constexpr int kL  = 13;
constexpr int kH  = 256, kW = 256;
constexpr int kH2 = 128, kW2 = 128;
constexpr int kHW = kH2 * kW2;          // 16384
constexpr int kN  = kB * kHW;           // 32768 : GEMM N dimension
// padded activation planes (2-pixel zero halo covers both pad=1 and pad=2 convs)
constexpr int kPB  = 2;
constexpr int kWp  = kW2 + 2 * kPB;     // 132
constexpr int kHp  = kH2 + 2 * kPB;     // 132
constexpr int kPHW = kWp * kHp;         // 17424
constexpr int kINT = kPB * kWp + kPB;   // interior (0,0) offset = 266
}

__device__ __forceinline__ float leaky_(float x) { return x >= 0.f ? x : 0.2f * x; }

// ------------------------------------------------------------------
// f32 -> f16 weight conversion with zero tail padding
// ------------------------------------------------------------------
__global__ void k_cvt_pad(const float* __restrict__ src, _Float16* __restrict__ dst,
                          int n_src, int n_total) {
  int i = blockIdx.x * blockDim.x + threadIdx.x;
  if (i < n_total) dst[i] = (i < n_src) ? (_Float16)src[i] : (_Float16)0.f;
}

__global__ void k_fill_f32(float* __restrict__ p, int n) {
  int i = blockIdx.x * blockDim.x + threadIdx.x;
  if (i < n) p[i] = 0.f;
}
__global__ void k_fill_f16(_Float16* __restrict__ p, int n) {
  int i = blockIdx.x * blockDim.x + threadIdx.x;
  if (i < n) p[i] = (_Float16)0.f;
}

// ------------------------------------------------------------------
// Stem: (B, T*1, 256,256) --3x3 s2 p1--> leaky -> f16 padded sequence
// xseqInt points at the interior (0,0) of padded [b][t][c][132][132] planes.
// ------------------------------------------------------------------
__global__ void k_stem(const float* __restrict__ x, const float* __restrict__ w,
                       const float* __restrict__ bias, _Float16* __restrict__ xseqInt) {
  int idx = blockIdx.x * blockDim.x + threadIdx.x;        // B*384*HW
  if (idx >= kB * 384 * kHW) return;
  int p  = idx % kHW;
  int ch = (idx / kHW) % 384;
  int b  = idx / (kHW * 384);
  int y2 = p / kW2, x2 = p % kW2;
  float acc = bias[ch];
#pragma unroll
  for (int ci = 0; ci < kT; ++ci) {
    const float* xb = x + ((size_t)(b * kT + ci)) * kH * kW;
    const float* wb = w + ((size_t)ch * kT + ci) * 9;
#pragma unroll
    for (int r = 0; r < 3; ++r) {
      int yy = 2 * y2 + r - 1;
#pragma unroll
      for (int s = 0; s < 3; ++s) {
        int xx = 2 * x2 + s - 1;
        bool ok = (yy >= 0) && (yy < kH) && (xx >= 0) && (xx < kW);
        int yc = ok ? yy : 0, xc = ok ? xx : 0;
        float v = xb[yc * kW + xc];
        acc += (ok ? v : 0.f) * wb[r * 3 + s];
      }
    }
  }
  acc = leaky_(acc);
  int t = ch / kC, c = ch % kC;
  xseqInt[(((size_t)(b * kT + t)) * kC + c) * kPHW + y2 * kWp + x2] = (_Float16)acc;
}

// ------------------------------------------------------------------
// Explicit im2col into n-major matrix:  Bm[n][k], k = c*R*S + r*S + s.
// Source planes are zero-halo padded -> no bounds checks.
// One block = 256 consecutive k for a single n: writes fully coalesced.
// ------------------------------------------------------------------
template <int R, int S, int PAD>
__global__ void k_im2col(const _Float16* __restrict__ Xint, size_t xBatchStride,
                         _Float16* __restrict__ Bm, int K) {
  constexpr int RS = R * S;
  int n = blockIdx.y;
  int k = blockIdx.x * blockDim.x + threadIdx.x;
  if (k >= K) return;
  int b = n / kHW, p = n % kHW;
  int y = p / kW2, x = p % kW2;
  int c = k / RS;
  int rs = k - c * RS;
  int r = rs / S;
  int s = rs - r * S;
  Bm[(size_t)n * K + k] =
      Xint[(size_t)b * xBatchStride + (size_t)c * kPHW + (y + r - PAD) * kWp + (x + s - PAD)];
}

// ------------------------------------------------------------------
// WMMA GEMM:  out[M,N] = act( A[M,K] * Bn[N,K]^T + bias [+ addIn] )
// A  : f16 row-major [Mpad, K]
// Bn : f16 n-major   [N, K]   (explicit im2col / warp output)
// Block = 128 threads (4 waves); each wave owns MT 16x16 M-tiles sharing one
// B fragment per K-chunk (B is read exactly once per GEMM when MT*16 == M).
// K is compile-time so per-tile A offsets (m*16*K) fold to immediate offsets.
// Fragment layouts per CDNA5 wave32 WMMA striping:
//   A (16x32 f16): row = lane&15; K offsets {h*8+j, 16+h*8+j}, h = lane>>4
//   B (32x16 f16): col = lane&15; K offsets {h*8+j, 16+h*8+j} (same as A)
//   C/D (16x16 f32): col = lane&15; row = i + h*8
// ------------------------------------------------------------------
template <int ACT, int MT, int K, bool HAS_ADD, bool HAS_MIR>
__global__ void __launch_bounds__(128)
k_gemm(const _Float16* __restrict__ Ah,
       const _Float16* __restrict__ Bn,
       const float* __restrict__ bias,
       float* __restrict__ out,
       const float* __restrict__ addIn,     // used iff HAS_ADD
       _Float16* __restrict__ mirrorInt,    // used iff HAS_MIR (padded-interior)
       int Mout) {
  const int lane = threadIdx.x & 31;
  const int wave = threadIdx.x >> 5;
  const int half = lane >> 4;
  const int l16  = lane & 15;
  const int tileN = blockIdx.x * 64 + wave * 16;
  const int tileM = blockIdx.y * (16 * MT);

  const int n = tileN + l16;         // exact: N is a multiple of 64
  const int b = n / kHW;
  const int p = n % kHW;
  const int y = p / kW2;
  const int x = p % kW2;

  v8f acc[MT];
#pragma unroll
  for (int m = 0; m < MT; ++m)
#pragma unroll
    for (int i = 0; i < 8; ++i) {
      int row = tileM + m * 16 + i + half * 8;
      float bv = 0.f;
      if (row < Mout) bv = bias[row];
      acc[m][i] = bv;
    }

  const _Float16* Abase = Ah + (size_t)(tileM + l16) * K;    // row for m=0
  const _Float16* Brow  = Bn + (size_t)n * K;

  for (int k0 = 0; k0 < K; k0 += 32) {
    // ---- B fragment: K {h*8+j, 16+h*8+j} -> two aligned b128 loads ----
    v8h blo = *reinterpret_cast<const v8h*>(Brow + k0 + half * 8);
    v8h bhi = *reinterpret_cast<const v8h*>(Brow + k0 + 16 + half * 8);
    v16h bf;
#pragma unroll
    for (int j = 0; j < 8; ++j) { bf[j] = blo[j]; bf[j + 8] = bhi[j]; }
    // ---- MT A fragments (immediate offsets m*16*K) + MT WMMAs sharing bf ----
#pragma unroll
    for (int m = 0; m < MT; ++m) {
      v8h alo = *reinterpret_cast<const v8h*>(Abase + (size_t)m * 16 * K + k0 + half * 8);
      v8h ahi = *reinterpret_cast<const v8h*>(Abase + (size_t)m * 16 * K + k0 + 16 + half * 8);
      v16h a;
#pragma unroll
      for (int j = 0; j < 8; ++j) { a[j] = alo[j]; a[j + 8] = ahi[j]; }
      acc[m] = __builtin_amdgcn_wmma_f32_16x16x32_f16(
          /*neg_a=*/false, a, /*neg_b=*/false, bf,
          /*c_mod=*/(short)0, acc[m], /*reuse_a=*/false, /*reuse_b=*/false);
    }
  }

#pragma unroll
  for (int m = 0; m < MT; ++m)
#pragma unroll
    for (int i = 0; i < 8; ++i) {
      int row = tileM + m * 16 + i + half * 8;
      if (row < Mout) {
        size_t o = ((size_t)b * Mout + row) * kHW + p;
        float v = acc[m][i];
        if (HAS_ADD) v += addIn[o];
        if (ACT == 1) v = leaky_(v);
        out[o] = v;
        if (HAS_MIR)
          mirrorInt[((size_t)b * Mout + row) * kPHW + y * kWp + x] = (_Float16)v;
      }
    }
}

// ------------------------------------------------------------------
// Bilinear warp of hprev by 13 flow fields; writes n-major B matrix for
// the 832->192 1x1 GEMM:  Bret[n][k], k = c*13 + l  (w_ret layout).
// ------------------------------------------------------------------
__global__ void k_warp(const float* __restrict__ flows, const float* __restrict__ hprev,
                       _Float16* __restrict__ Bret) {
  int idx = blockIdx.x * blockDim.x + threadIdx.x;        // B*L*HW
  if (idx >= kB * kL * kHW) return;
  int p = idx % kHW;
  int l = (idx / kHW) % kL;
  int b = idx / (kHW * kL);
  int y = p / kW2, x = p % kW2;

  float u = flows[((size_t)b * 2 * kL + 2 * l) * kHW + p];
  float v = flows[((size_t)b * 2 * kL + 2 * l + 1) * kHW + p];
  float sx = (float)x - u;
  float sy = (float)y - v;
  float x0f = floorf(sx), y0f = floorf(sy);
  int x0 = (int)x0f, y0 = (int)y0f;
  float wx1 = sx - x0f, wx0 = 1.f - wx1;
  float wy1 = sy - y0f, wy0 = 1.f - wy1;

  bool vx0 = (x0 >= 0) && (x0 <= kW2 - 1);
  bool vx1 = (x0 + 1 >= 0) && (x0 + 1 <= kW2 - 1);
  bool vy0 = (y0 >= 0) && (y0 <= kH2 - 1);
  bool vy1 = (y0 + 1 >= 0) && (y0 + 1 <= kH2 - 1);
  int x0c = min(max(x0, 0), kW2 - 1);
  int x1c = min(max(x0 + 1, 0), kW2 - 1);
  int y0c = min(max(y0, 0), kH2 - 1);
  int y1c = min(max(y0 + 1, 0), kH2 - 1);
  float w00 = wx0 * wy0 * ((vx0 && vy0) ? 1.f : 0.f);
  float w10 = wx1 * wy0 * ((vx1 && vy0) ? 1.f : 0.f);
  float w01 = wx0 * wy1 * ((vx0 && vy1) ? 1.f : 0.f);
  float w11 = wx1 * wy1 * ((vx1 && vy1) ? 1.f : 0.f);
  int o00 = y0c * kW2 + x0c, o10 = y0c * kW2 + x1c;
  int o01 = y1c * kW2 + x0c, o11 = y1c * kW2 + x1c;

  const float* hb = hprev + (size_t)b * kC * kHW;
  _Float16* brow = Bret + (size_t)(b * kHW + p) * (kC * kL) + l;
  for (int c = 0; c < kC; ++c) {
    const float* hc = hb + (size_t)c * kHW;
    float g = hc[o00] * w00 + hc[o10] * w10 + hc[o01] * w01 + hc[o11] * w11;
    brow[c * kL] = (_Float16)g;
  }
}

// ------------------------------------------------------------------
// GRU pointwise update. Writes new hidden (f32 flat + f16 padded mirror)
// and the d_out slice for this timestep.
// ------------------------------------------------------------------
__global__ void k_gru(const float* __restrict__ i2h, const float* __restrict__ h2h,
                      float* __restrict__ hprev, _Float16* __restrict__ hmirInt,
                      float* __restrict__ outBase /* d_out + t*64*HW */) {
  int idx = blockIdx.x * blockDim.x + threadIdx.x;        // B*64*HW
  if (idx >= kB * kC * kHW) return;
  int p = idx % kHW;
  int c = (idx / kHW) % kC;
  int b = idx / (kHW * kC);
  int y = p / kW2, x = p % kW2;
  size_t base = (size_t)b * 3 * kC * kHW;
  size_t o    = (size_t)c * kHW + p;
  float ir = i2h[base + o];
  float iu = i2h[base + (size_t)kC * kHW + o];
  float im = i2h[base + (size_t)2 * kC * kHW + o];
  float hr = h2h[base + o];
  float hu = h2h[base + (size_t)kC * kHW + o];
  float hm = h2h[base + (size_t)2 * kC * kHW + o];
  float r = 1.f / (1.f + expf(-(ir + hr)));
  float z = 1.f / (1.f + expf(-(iu + hu)));
  float m = leaky_(im + r * hm);
  size_t ho = ((size_t)b * kC + c) * kHW + p;
  float hn = z * hprev[ho] + (1.f - z) * m;
  hprev[ho] = hn;
  hmirInt[((size_t)b * kC + c) * kPHW + y * kWp + x] = (_Float16)hn;
  outBase[(size_t)b * (kT * kC * kHW) + o] = hn;
}

// ------------------------------------------------------------------
extern "C" void kernel_launch(void* const* d_in, const int* in_sizes, int n_in,
                              void* d_out, int out_size, void* d_ws, size_t ws_size,
                              hipStream_t stream) {
  const float* x      = (const float*)d_in[0];
  const float* w_stem = (const float*)d_in[1];
  const float* b_stem = (const float*)d_in[2];
  const float* w_i2h  = (const float*)d_in[3];
  const float* b_i2h  = (const float*)d_in[4];
  const float* w_i2f  = (const float*)d_in[5];
  const float* b_i2f  = (const float*)d_in[6];
  const float* w_h2f  = (const float*)d_in[7];
  const float* b_h2f  = (const float*)d_in[8];
  const float* w_flow = (const float*)d_in[9];
  const float* b_flow = (const float*)d_in[10];
  const float* w_ret  = (const float*)d_in[11];
  const float* b_ret  = (const float*)d_in[12];
  float* out = (float*)d_out;

  char* wp = (char*)d_ws;
  auto carve = [&](size_t bytes) -> void* {
    void* r = (void*)wp;
    wp += (bytes + 255) & ~(size_t)255;
    return r;
  };
  _Float16* AhI2H   = (_Float16*)carve((size_t)192 * 576 * 2);
  _Float16* AhI2F   = (_Float16*)carve((size_t)32 * 1600 * 2);
  _Float16* AhH2F   = (_Float16*)carve((size_t)32 * 1600 * 2);
  _Float16* AhFLOW  = (_Float16*)carve((size_t)32 * 800 * 2);    // 26 rows + 6 zero pad
  _Float16* AhRET   = (_Float16*)carve((size_t)192 * 832 * 2);
  _Float16* XseqPad = (_Float16*)carve((size_t)kB * kT * kC * kPHW * 2);
  _Float16* HhPad   = (_Float16*)carve((size_t)kB * kC * kPHW * 2);
  _Float16* F1Pad   = (_Float16*)carve((size_t)kB * 32 * kPHW * 2);
  float*    Hf32    = (float*)   carve((size_t)kB * kC * kHW * 4);
  float*    I2H     = (float*)   carve((size_t)kB * 192 * kHW * 4);
  float*    F1      = (float*)   carve((size_t)kB * 32 * kHW * 4);
  float*    Flows   = (float*)   carve((size_t)kB * 26 * kHW * 4);
  float*    H2H     = (float*)   carve((size_t)kB * 192 * kHW * 4);
  // one big n-major B matrix, serially reused by all five GEMMs per step
  _Float16* BmBig   = (_Float16*)carve((size_t)kN * 1600 * 2);   // 104.8 MB

  _Float16* XseqInt = XseqPad + kINT;
  _Float16* HhInt   = HhPad + kINT;
  _Float16* F1Int   = F1Pad + kINT;

  auto cvt = [&](const float* s, _Float16* d, int ns, int nt) {
    k_cvt_pad<<<(nt + 255) / 256, 256, 0, stream>>>(s, d, ns, nt);
  };
  cvt(w_i2h,  AhI2H,  192 * 576,  192 * 576);
  cvt(w_i2f,  AhI2F,  32 * 1600,  32 * 1600);
  cvt(w_h2f,  AhH2F,  32 * 1600,  32 * 1600);
  cvt(w_flow, AhFLOW, 26 * 800,   32 * 800);
  cvt(w_ret,  AhRET,  192 * 832,  192 * 832);

  // zero halo'd activation buffers (interiors rewritten each use; halos stay 0)
  {
    int n1 = kB * kT * kC * kPHW;
    k_fill_f16<<<(n1 + 255) / 256, 256, 0, stream>>>(XseqPad, n1);
    int n2 = kB * kC * kPHW;
    k_fill_f16<<<(n2 + 255) / 256, 256, 0, stream>>>(HhPad, n2);
    int n3 = kB * 32 * kPHW;
    k_fill_f16<<<(n3 + 255) / 256, 256, 0, stream>>>(F1Pad, n3);
    int n4 = kB * kC * kHW;
    k_fill_f32<<<(n4 + 255) / 256, 256, 0, stream>>>(Hf32, n4);
  }

  // stem (K=54 direct VALU conv, ~1.4 GFLOP total)
  {
    int total = kB * 384 * kHW;
    k_stem<<<(total + 255) / 256, 256, 0, stream>>>(x, w_stem, b_stem, XseqInt);
  }

  const dim3 blk(128);
  for (int t = 0; t < kT; ++t) {
    const _Float16* xt = XseqInt + (size_t)t * kC * kPHW;     // b-stride = T*64*PHW
    const size_t xtBS = (size_t)kT * kC * kPHW;

    // i2h = conv3x3(xt)                        (M=192, MT=12 -> grid.y=1, B read once)
    k_im2col<3, 3, 1><<<dim3((576 + 255) / 256, kN), 256, 0, stream>>>(xt, xtBS, BmBig, 576);
    k_gemm<0, 12, 576, false, false><<<dim3(kN / 64, 1), blk, 0, stream>>>(
        AhI2H, BmBig, b_i2h, I2H, nullptr, nullptr, 192);

    // F1 = conv5x5_i2f(xt)  (raw)              (M=32, MT=2 -> grid.y=1)
    k_im2col<5, 5, 2><<<dim3((1600 + 255) / 256, kN), 256, 0, stream>>>(xt, xtBS, BmBig, 1600);
    k_gemm<0, 2, 1600, false, false><<<dim3(kN / 64, 1), blk, 0, stream>>>(
        AhI2F, BmBig, b_i2f, F1, nullptr, nullptr, 32);

    // F1 = leaky(F1 + conv5x5_h2f(h)), f16 padded mirror
    k_im2col<5, 5, 2><<<dim3((1600 + 255) / 256, kN), 256, 0, stream>>>(
        HhInt, (size_t)kC * kPHW, BmBig, 1600);
    k_gemm<1, 2, 1600, true, true><<<dim3(kN / 64, 1), blk, 0, stream>>>(
        AhH2F, BmBig, b_h2f, F1, F1, F1Int, 32);

    // flows = conv5x5(F1) (26 rows of 32)
    k_im2col<5, 5, 2><<<dim3((800 + 255) / 256, kN), 256, 0, stream>>>(
        F1Int, (size_t)32 * kPHW, BmBig, 800);
    k_gemm<0, 2, 800, false, false><<<dim3(kN / 64, 1), blk, 0, stream>>>(
        AhFLOW, BmBig, b_flow, Flows, nullptr, nullptr, 26);

    // warp hidden by flows -> n-major f16 [N][832]
    {
      int total = kB * kL * kHW;
      k_warp<<<(total + 255) / 256, 256, 0, stream>>>(Flows, Hf32, BmBig);
    }
    // h2h = 1x1 conv (832 -> 192)              (M=192, MT=12 -> grid.y=1, B read once)
    k_gemm<0, 12, 832, false, false><<<dim3(kN / 64, 1), blk, 0, stream>>>(
        AhRET, BmBig, b_ret, H2H, nullptr, nullptr, 192);

    // GRU update; writes d_out[:, t] and hidden (f32 + padded f16)
    {
      int total = kB * kC * kHW;
      k_gru<<<(total + 255) / 256, 256, 0, stream>>>(I2H, H2H, Hf32, HhInt,
                                                     out + (size_t)t * kC * kHW);
    }
  }
}
